// TGNMemoryModule_83502754168895
// MI455X (gfx1250) — matrix operand order repaired and checked by reading, not compiled, not run on hardware
//
#include <hip/hip_runtime.h>
#include <math.h>

// CDNA5 wave32 WMMA types
typedef __attribute__((ext_vector_type(16))) __bf16 v16bf;
typedef __attribute__((ext_vector_type(8)))  float  v8f;

#define MEMD  128
#define RAWD  484
#define RAWP  512   // K padded to 512 for the message GEMM
#define TIMED 100
#define GATES 384   // 3 * MEMD

__device__ __forceinline__ v8f wmma_bf16(v16bf a, v16bf b, v8f c) {
  // emits v_wmma_f32_16x16x32_bf16
  return __builtin_amdgcn_wmma_f32_16x16x32_bf16(false, a, false, b, (short)0, c,
                                                 false, false);
}

__device__ __forceinline__ float fast_sigmoid(float x) {
  return 1.0f / (1.0f + __expf(-x));
}
__device__ __forceinline__ float fast_tanh(float x) {
  return 1.0f - 2.0f / (__expf(2.0f * x) + 1.0f);
}

// A-fragment (16-bit A 16x32 layout): lane holds K = k0..k0+7 and k0+16..k0+23.
// p+k0 is 32B-aligned. Four b128 loads + 8 packed cvts.
__device__ __forceinline__ v16bf a_frag16(const float* __restrict__ p, int k0) {
  const float4 x0 = *(const float4*)(p + k0);
  const float4 x1 = *(const float4*)(p + k0 + 4);
  const float4 x2 = *(const float4*)(p + k0 + 16);
  const float4 x3 = *(const float4*)(p + k0 + 20);
  v16bf a;
  a[0]=(__bf16)x0.x;  a[1]=(__bf16)x0.y;  a[2]=(__bf16)x0.z;  a[3]=(__bf16)x0.w;
  a[4]=(__bf16)x1.x;  a[5]=(__bf16)x1.y;  a[6]=(__bf16)x1.z;  a[7]=(__bf16)x1.w;
  a[8]=(__bf16)x2.x;  a[9]=(__bf16)x2.y;  a[10]=(__bf16)x2.z; a[11]=(__bf16)x2.w;
  a[12]=(__bf16)x3.x; a[13]=(__bf16)x3.y; a[14]=(__bf16)x3.z; a[15]=(__bf16)x3.w;
  return a;
}

__device__ __forceinline__ v16bf a_frag16_scaled(const float* __restrict__ p,
                                                 int k0, float s) {
  const float4 x0 = *(const float4*)(p + k0);
  const float4 x1 = *(const float4*)(p + k0 + 4);
  const float4 x2 = *(const float4*)(p + k0 + 16);
  const float4 x3 = *(const float4*)(p + k0 + 20);
  v16bf a;
  a[0]=(__bf16)(x0.x*s);  a[1]=(__bf16)(x0.y*s);  a[2]=(__bf16)(x0.z*s);  a[3]=(__bf16)(x0.w*s);
  a[4]=(__bf16)(x1.x*s);  a[5]=(__bf16)(x1.y*s);  a[6]=(__bf16)(x1.z*s);  a[7]=(__bf16)(x1.w*s);
  a[8]=(__bf16)(x2.x*s);  a[9]=(__bf16)(x2.y*s);  a[10]=(__bf16)(x2.z*s); a[11]=(__bf16)(x2.w*s);
  a[12]=(__bf16)(x3.x*s); a[13]=(__bf16)(x3.y*s); a[14]=(__bf16)(x3.z*s); a[15]=(__bf16)(x3.w*s);
  return a;
}

// B-fragment from pre-converted bf16 weights: two 16B loads (K k0..k0+7 and
// k0+16..k0+23), zero conversions. p must be 16B-aligned (all our offsets are).
__device__ __forceinline__ v16bf b_frag_bf16(const __bf16* __restrict__ p) {
  typedef __attribute__((ext_vector_type(8))) __bf16 v8bf;
  const v8bf lo = *(const v8bf*)p;
  const v8bf hi = *(const v8bf*)(p + 16);
  return __builtin_shufflevector(lo, hi, 0, 1, 2, 3, 4, 5, 6, 7,
                                 8, 9, 10, 11, 12, 13, 14, 15);
}

__global__ void zero_kernel(float* __restrict__ p, long n) {
  long n4 = n >> 2;
  float4* p4 = (float4*)p;
  const float4 z4 = make_float4(0.f, 0.f, 0.f, 0.f);
  long i = (long)blockIdx.x * blockDim.x + threadIdx.x;
  long stride = (long)gridDim.x * blockDim.x;
  for (long t = i; t < n4; t += stride) p4[t] = z4;
  for (long t = (n4 << 2) + i; t < n; t += stride) p[t] = 0.0f;
}

// One-shot per-launch weight conversion (f32 -> bf16), deterministic.
__global__ void cvt_bf16_kernel(const float* __restrict__ src,
                                __bf16* __restrict__ dst, int n) {
  int i = blockIdx.x * blockDim.x + threadIdx.x;
  if (i < n) dst[i] = (__bf16)src[i];
}

// msg_W [128 x 484] -> bf16 [128 x 512], K-padded with zeros.
__global__ void cvt_pad_msgw_kernel(const float* __restrict__ src,
                                    __bf16* __restrict__ dst) {
  int i = blockIdx.x * blockDim.x + threadIdx.x;
  if (i >= MEMD * RAWP) return;
  int row = i >> 9;
  int k   = i & (RAWP - 1);
  dst[i] = (k < RAWD) ? (__bf16)src[row * RAWD + k] : (__bf16)0.0f;
}

// Kernel 1: per-edge message GEMM (bf16 WMMA, fp32 accum) fused with
// time-encoding build and atomic scatter into node sums/counts.
// One wave = 16 edges x 128 message dims; K = 484 padded to 512.
__global__ void msg_scatter_kernel(
    const float* __restrict__ memory, const float* __restrict__ last_update,
    const float* __restrict__ timestamps, const float* __restrict__ edge_features,
    const __bf16* __restrict__ msg_Wb, const float* __restrict__ msg_b,
    const float* __restrict__ time_w, const float* __restrict__ time_b,
    const int* __restrict__ src, const int* __restrict__ dst,
    float* __restrict__ sums, float* __restrict__ counts, int n_tiles)
{
  const int wave = threadIdx.x >> 5;
  const int lane = threadIdx.x & 31;
  const int tile = blockIdx.x * (blockDim.x >> 5) + wave;
  if (tile >= n_tiles) return;           // wave-uniform: EXEC stays all-ones

  const int e0   = tile * 16;
  const int m    = lane & 15;
  const int e    = e0 + m;
  const int se   = src[e];
  const int de   = dst[e];
  const int half = (lane < 16) ? 0 : 8;
  const int kA   = half;                 // A lane K-base: 0 / 8
  const int kB   = half;                 // B lane K-base: 0 / 8
  const float dtv = timestamps[e] - last_update[se];

  const float*  __restrict__ srow  = memory + (size_t)se * MEMD;
  const float*  __restrict__ drow  = memory + (size_t)de * MEMD;
  const float*  __restrict__ erow  = edge_features + (size_t)e * MEMD;
  const __bf16* __restrict__ wbase = msg_Wb + (size_t)m * RAWP + kB;

  v8f acc[8];
  v8f zero = {};
  #pragma unroll
  for (int i = 0; i < 8; ++i) acc[i] = zero;

  // Regions 0..2: src_mem / dst_mem / edge_feat (contiguous rows, vector loads)
  #pragma unroll
  for (int rg = 0; rg < 3; ++rg) {
    const float* arow = (rg == 0) ? srow : (rg == 1) ? drow : erow;
    #pragma unroll
    for (int kq = 0; kq < 4; ++kq) {
      const int koff = rg * 128 + kq * 32;
      // Batch all 8 B-fragment loads first: one long load clause, then the
      // WMMA chain drains it with descending s_wait_loadcnt (load/WMMA overlap)
      v16bf bfr[8];
      #pragma unroll
      for (int nt = 0; nt < 8; ++nt)
        bfr[nt] = b_frag_bf16(wbase + (size_t)nt * 16 * RAWP + koff);
      v16bf a = a_frag16(arow, kq * 32 + kA);
      #pragma unroll
      for (int nt = 0; nt < 8; ++nt)
        acc[nt] = wmma_bf16(a, bfr[nt], acc[nt]);
    }
  }

  // Region 3: time encoding (K 384..483; weights zero-padded to 512 so no
  // tail guard needed on the B side). |dtv*w| <= ~2 -> v_cos_f32 fast path.
  #pragma unroll
  for (int kq = 0; kq < 4; ++kq) {
    const int koff = 384 + kq * 32;
    v16bf bfr[8];
    #pragma unroll
    for (int nt = 0; nt < 8; ++nt)
      bfr[nt] = b_frag_bf16(wbase + (size_t)nt * 16 * RAWP + koff);
    v16bf a;
    #pragma unroll
    for (int j = 0; j < 16; ++j) {
      const int kt = kq * 32 + kA + (j < 8 ? j : 8 + j);
      float v = 0.0f;
      if (kt < TIMED) v = __cosf(dtv * time_w[kt] + time_b[kt]);
      a[j] = (__bf16)v;
    }
    #pragma unroll
    for (int nt = 0; nt < 8; ++nt)
      acc[nt] = wmma_bf16(a, bfr[nt], acc[nt]);
  }

  // D layout: element v of lane = (M = v + half, N = lane&15)
  int seM[8], deM[8];
  #pragma unroll
  for (int v = 0; v < 8; ++v) {
    seM[v] = __shfl(se, v + half, 32);
    deM[v] = __shfl(de, v + half, 32);
  }
  #pragma unroll
  for (int nt = 0; nt < 8; ++nt) {
    const int col = nt * 16 + m;
    const float mb = msg_b[col];
    #pragma unroll
    for (int v = 0; v < 8; ++v) {
      const float val = acc[nt][v] + mb;
      atomicAdd(&sums[(size_t)seM[v] * MEMD + col], val);
      atomicAdd(&sums[(size_t)deM[v] * MEMD + col], val);
    }
  }
  if (lane < 16) {
    atomicAdd(&counts[se], 1.0f);
    atomicAdd(&counts[de], 1.0f);
  }
}

// Kernel 2: fused GRU. One wave = 16 nodes; gi = agg@W_ih^T, gh = mem@W_hh^T
// via bf16 WMMA, then gates + masked writeback + last_update.
__global__ void gru_kernel(
    const float* __restrict__ memory, const float* __restrict__ last_update,
    const float* __restrict__ timestamps,
    const __bf16* __restrict__ Wihb, const __bf16* __restrict__ Whhb,
    const float* __restrict__ b_ih, const float* __restrict__ b_hh,
    const float* __restrict__ sums, const float* __restrict__ counts,
    float* __restrict__ out_mem, float* __restrict__ out_lu, int n_tiles)
{
  const int wave = threadIdx.x >> 5;
  const int lane = threadIdx.x & 31;
  const int tile = blockIdx.x * (blockDim.x >> 5) + wave;
  if (tile >= n_tiles) return;           // wave-uniform

  const int n0   = tile * 16;
  const int m    = lane & 15;
  const int node = n0 + m;
  const int half = (lane < 16) ? 0 : 8;
  const int kA   = half;
  const int kB   = half;
  const float cnt  = counts[node];
  const float invc = 1.0f / fmaxf(cnt, 1.0f);
  const float ts0  = timestamps[0];

  const float* __restrict__ arow = sums + (size_t)node * MEMD;
  const float* __restrict__ mrow = memory + (size_t)node * MEMD;

  v16bf aAgg[4], aMem[4];
  #pragma unroll
  for (int kc = 0; kc < 4; ++kc) {
    aAgg[kc] = a_frag16_scaled(arow, kc * 32 + kA, invc);
    aMem[kc] = a_frag16(mrow, kc * 32 + kA);
  }

  float cM[8];
  #pragma unroll
  for (int v = 0; v < 8; ++v) cM[v] = __shfl(cnt, v + half, 32);

  v8f zero = {};
  #pragma unroll 1
  for (int nt = 0; nt < 8; ++nt) {
    const int col = nt * 16 + m;         // output column, lane-constant
    v8f gi[3], gh[3];
    #pragma unroll
    for (int g = 0; g < 3; ++g) { gi[g] = zero; gh[g] = zero; }

    #pragma unroll
    for (int kc = 0; kc < 4; ++kc) {
      // Batch the 6 weight-fragment loads, then issue the 6 WMMAs
      v16bf bi[3], bh[3];
      #pragma unroll
      for (int g = 0; g < 3; ++g) {
        const size_t woff = (size_t)(g * MEMD + col) * MEMD + kc * 32 + kB;
        bi[g] = b_frag_bf16(Wihb + woff);
        bh[g] = b_frag_bf16(Whhb + woff);
      }
      #pragma unroll
      for (int g = 0; g < 3; ++g) {      // r, z, n gate blocks of 128 rows
        gi[g] = wmma_bf16(aAgg[kc], bi[g], gi[g]);
        gh[g] = wmma_bf16(aMem[kc], bh[g], gh[g]);
      }
    }

    const float bir = b_ih[col], biz = b_ih[MEMD + col], bin = b_ih[2 * MEMD + col];
    const float bhr = b_hh[col], bhz = b_hh[MEMD + col], bhn = b_hh[2 * MEMD + col];

    #pragma unroll
    for (int v = 0; v < 8; ++v) {
      const int nd = n0 + v + half;
      const float r  = fast_sigmoid(gi[0][v] + bir + gh[0][v] + bhr);
      const float z  = fast_sigmoid(gi[1][v] + biz + gh[1][v] + bhz);
      const float nn = fast_tanh(gi[2][v] + bin + r * (gh[2][v] + bhn));
      const float hp = memory[(size_t)nd * MEMD + col];
      const float h  = (1.0f - z) * nn + z * hp;
      out_mem[(size_t)nd * MEMD + col] = (cM[v] > 0.0f) ? h : hp;
    }
  }

  if (lane < 16) {
    out_lu[node] = (cnt > 0.0f) ? ts0 : last_update[node];
  }
}

extern "C" void kernel_launch(void* const* d_in, const int* in_sizes, int n_in,
                              void* d_out, int out_size, void* d_ws, size_t ws_size,
                              hipStream_t stream) {
  const float* memory        = (const float*)d_in[0];
  const float* last_update   = (const float*)d_in[1];
  const float* timestamps    = (const float*)d_in[2];
  const float* edge_features = (const float*)d_in[3];
  const float* msg_W         = (const float*)d_in[4];
  const float* msg_b         = (const float*)d_in[5];
  const float* W_ih          = (const float*)d_in[6];
  const float* W_hh          = (const float*)d_in[7];
  const float* b_ih          = (const float*)d_in[8];
  const float* b_hh          = (const float*)d_in[9];
  const float* time_w        = (const float*)d_in[10];
  const float* time_b        = (const float*)d_in[11];
  const int*   src           = (const int*)d_in[12];
  const int*   dst           = (const int*)d_in[13];

  const int N = in_sizes[0] / MEMD;   // 200000
  const int E = in_sizes[2];          // 100000

  float* out_mem = (float*)d_out;
  float* out_lu  = out_mem + (size_t)N * MEMD;

  // Workspace layout: f32 sums [N,128] | f32 counts [N] | bf16 weights
  float* sums   = (float*)d_ws;
  float* counts = sums + (size_t)N * MEMD;
  size_t foff   = ((size_t)N * MEMD + N + 63) & ~(size_t)63;  // 256B align
  __bf16* msg_Wb = (__bf16*)(sums + foff);                    // [128 x 512]
  __bf16* Wihb   = msg_Wb + (size_t)MEMD * RAWP;              // [384 x 128]
  __bf16* Whhb   = Wihb + (size_t)GATES * MEMD;               // [384 x 128]

  // Per-launch weight conversion (deterministic, ~164K elements total)
  cvt_pad_msgw_kernel<<<(MEMD * RAWP + 255) / 256, 256, 0, stream>>>(msg_W, msg_Wb);
  cvt_bf16_kernel<<<(GATES * MEMD + 255) / 256, 256, 0, stream>>>(W_ih, Wihb,
                                                                  GATES * MEMD);
  cvt_bf16_kernel<<<(GATES * MEMD + 255) / 256, 256, 0, stream>>>(W_hh, Whhb,
                                                                  GATES * MEMD);

  // Zero scratch every call (graph-replay deterministic)
  long zn = (long)N * MEMD + N;
  int  zb = (int)((zn / 4 + 255) / 256);
  zero_kernel<<<zb, 256, 0, stream>>>(sums, zn);

  // Messages + scatter: one wave per 16-edge tile, 4 waves per block
  int etiles = (E + 15) / 16;
  int eblk   = (etiles + 3) / 4;
  msg_scatter_kernel<<<eblk, 128, 0, stream>>>(
      memory, last_update, timestamps, edge_features, msg_Wb, msg_b,
      time_w, time_b, src, dst, sums, counts, etiles);

  // GRU: one wave per 16-node tile, 4 waves per block
  int ntiles = (N + 15) / 16;
  int nblk   = (ntiles + 3) / 4;
  gru_kernel<<<nblk, 128, 0, stream>>>(
      memory, last_update, timestamps, Wihb, Whhb, b_ih, b_hh,
      sums, counts, out_mem, out_lu, ntiles);
}
